// RBFNet_17892833755792
// MI455X (gfx1250) — matrix operand order repaired and checked by reading, not compiled, not run on hardware
//
#include <hip/hip_runtime.h>

typedef __attribute__((ext_vector_type(16))) _Float16 v16h;
typedef __attribute__((ext_vector_type(4)))  _Float16 v4h;
typedef __attribute__((ext_vector_type(8)))  float    v8f;

constexpr int F = 256;       // IN_FEATURES
constexpr int C = 4096;      // NUM_CENTERS
constexpr int B = 16384;     // BATCH
constexpr int CSPLIT = 4;    // split of center dim across waves
constexpr int CCHUNK = C / CSPLIT;   // 1024 centers per wave
constexpr int WPB = 8;       // waves per block (256 threads)

// ---------------------------------------------------------------------------
// Prep: centers f32 -> f16 (for WMMA), plus ||c||^2 per center row.
// One block (64 threads) per center row; each thread converts 4 elements.
// ---------------------------------------------------------------------------
__global__ __launch_bounds__(64) void rbf_prep(const float* __restrict__ centers,
                                               _Float16* __restrict__ ch,
                                               float* __restrict__ csq) {
    const int row = blockIdx.x;
    const int t = threadIdx.x;                    // 0..63
    const float4 f = *(const float4*)(centers + (size_t)row * F + t * 4);
    v4h o;
    o[0] = (_Float16)f.x; o[1] = (_Float16)f.y;
    o[2] = (_Float16)f.z; o[3] = (_Float16)f.w;
    *(v4h*)(ch + (size_t)row * F + t * 4) = o;
    float s = f.x * f.x + f.y * f.y + f.z * f.z + f.w * f.w;
    #pragma unroll
    for (int d = 1; d < 32; d <<= 1) s += __shfl_xor(s, d, 32);
    __shared__ float ls[2];
    if ((t & 31) == 0) ls[t >> 5] = s;
    __syncthreads();
    if (t == 0) csq[row] = ls[0] + ls[1];
}

// ---------------------------------------------------------------------------
// Main: per wave, one 16-row strip of x (A held in registers, K=256) swept
// against a 1024-center chunk in 16-center tiles. 8x v_wmma_f32_16x16x32_f16
// per tile, fused exp + w-weighted reduction in the epilogue.
// All 8 waves of a block share the SAME center chunk (different row strips)
// so B-fragment loads hit WGP$ instead of re-streaming L2 per wave.
// ---------------------------------------------------------------------------
__global__ __launch_bounds__(256) void rbf_main(const float* __restrict__ x,
                                                const _Float16* __restrict__ ch,
                                                const float* __restrict__ csq,
                                                const float* __restrict__ w,
                                                float* __restrict__ partial) {
    const int lane = threadIdx.x & 31;
    const int wave = threadIdx.x >> 5;
    const int chunk = blockIdx.x & (CSPLIT - 1);          // shared by whole block
    const int strip = (blockIdx.x >> 2) * WPB + wave;     // 0..1023
    const int row_base = strip << 4;
    const int half = lane >> 4;                   // wave32: two 16-lane halves
    const int rn = lane & 15;

    // ---- Load A fragments (16x256 f32 -> f16), accumulate ||x||^2 on the fly.
    // A 16-bit layout: lane(half,rn) holds 16 contiguous K = 16h..16h+15
    // per 32-K step (matches B).
    v16h a[8];
    float xsq = 0.f;
    {
        const float* xr = x + (size_t)(row_base + rn) * F + half * 16;
        #pragma unroll
        for (int k = 0; k < 8; ++k) {
            const float4 f0 = *(const float4*)(xr + 32 * k);
            const float4 f1 = *(const float4*)(xr + 32 * k + 4);
            const float4 f2 = *(const float4*)(xr + 32 * k + 8);
            const float4 f3 = *(const float4*)(xr + 32 * k + 12);
            float f[16] = {f0.x, f0.y, f0.z, f0.w, f1.x, f1.y, f1.z, f1.w,
                           f2.x, f2.y, f2.z, f2.w, f3.x, f3.y, f3.z, f3.w};
            #pragma unroll
            for (int e = 0; e < 16; ++e) {
                xsq += f[e] * f[e];
                a[k][e] = (_Float16)f[e];
            }
        }
    }
    // lanes rn and rn+16 each saw half the K range of row rn
    xsq += __shfl_xor(xsq, 16, 32);
    // acc[v] belongs to row (half*8 + v); lane (half*8+v) holds that row's xsq
    float xs2[8];
    #pragma unroll
    for (int v = 0; v < 8; ++v) xs2[v] = -0.5f * __shfl(xsq, half * 8 + v, 32);

    float racc[8] = {0.f, 0.f, 0.f, 0.f, 0.f, 0.f, 0.f, 0.f};
    const int cbase0 = chunk * CCHUNK;

    for (int ct = 0; ct < CCHUNK / 16; ++ct) {
        const int c = cbase0 + ct * 16 + rn;      // this lane's center column
        // B 16-bit layout: lane(half,n) holds 16 contiguous K = 16h..16h+15
        const _Float16* br = ch + (size_t)c * F + half * 16;
        v8f acc = {};
        #pragma unroll
        for (int k = 0; k < 8; ++k) {
            const v16h bfrag = *(const v16h*)(br + 32 * k);
            acc = __builtin_amdgcn_wmma_f32_16x16x32_f16(
                false, a[k], false, bfrag, (short)0, acc, false, false);
        }
        const float cs2 = -0.5f * csq[c];
        const float wv = w[c];
        #pragma unroll
        for (int v = 0; v < 8; ++v)
            racc[v] += wv * __expf(acc[v] + xs2[v] + cs2);  // exp(cross - ||x||²/2 - ||c||²/2)
    }

    // Reduce over the 16 column-lanes of each half (stays within a half).
    #pragma unroll
    for (int s = 1; s < 16; s <<= 1) {
        #pragma unroll
        for (int v = 0; v < 8; ++v) racc[v] += __shfl_xor(racc[v], s, 32);
    }
    if (rn == 0) {
        float* pp = partial + (size_t)chunk * B + row_base + half * 8;
        #pragma unroll
        for (int v = 0; v < 8; ++v) pp[v] = racc[v];
    }
}

// ---------------------------------------------------------------------------
// Final: out[i] = sigmoid(sum_chunks partial + bias)
// ---------------------------------------------------------------------------
__global__ __launch_bounds__(256) void rbf_final(const float* __restrict__ partial,
                                                 const float* __restrict__ bias,
                                                 float* __restrict__ out) {
    const int i = blockIdx.x * 256 + threadIdx.x;
    float s = bias[0];
    #pragma unroll
    for (int ch = 0; ch < CSPLIT; ++ch) s += partial[(size_t)ch * B + i];
    out[i] = 1.f / (1.f + __expf(-s));
}

extern "C" void kernel_launch(void* const* d_in, const int* in_sizes, int n_in,
                              void* d_out, int out_size, void* d_ws, size_t ws_size,
                              hipStream_t stream) {
    const float* x       = (const float*)d_in[0];   // [16384,256]
    const float* centers = (const float*)d_in[1];   // [4096,256]
    const float* w       = (const float*)d_in[2];   // [1,4096]
    const float* b       = (const float*)d_in[3];   // [1]
    float* out = (float*)d_out;                     // [16384]

    char* ws = (char*)d_ws;
    _Float16* ch  = (_Float16*)ws;                              // 4096*256*2 = 2 MB
    float* csq    = (float*)(ws + (size_t)C * F * 2);           // 16 KB
    float* part   = (float*)(ws + (size_t)C * F * 2 + C * 4);   // 4*16384*4 = 256 KB

    rbf_prep<<<C, 64, 0, stream>>>(centers, ch, csq);
    rbf_main<<<(B / 16 * CSPLIT) / WPB, WPB * 32, 0, stream>>>(x, ch, csq, w, part);
    rbf_final<<<B / 256, 256, 0, stream>>>(part, b, out);
}
